// TransMIL_77163382440362
// MI455X (gfx1250) — matrix operand order repaired and checked by reading, not compiled, not run on hardware
//
#include <hip/hip_runtime.h>

// ---------------------------------------------------------------------------
// TransMIL forward for MI455X (gfx1250, wave32, WMMA).
// Big GEMMs: bf16 inputs, f32 accumulate via v_wmma_f32_16x16x32_bf16.
//   - weight GEMMs (fc1/qkv/out-proj): 64x32 tile per wave (4x2 WMMA tiles)
//   - batched attention GEMMs: 32x32 tile per wave (2x2 WMMA tiles)
// Small numerically-sensitive ops (pinv Newton-Schulz, softmax, LN, depthwise
// convs) stay in fp32 VALU.
// ---------------------------------------------------------------------------

typedef __bf16 bf16;
typedef __attribute__((ext_vector_type(16))) __bf16 v16bf;
typedef __attribute__((ext_vector_type(8)))  float  v8f;

union FragU { v16bf v; uint4 q[2]; };

// Per-lane fragment base pointers (ISA VGPR layouts):
//  A (16x32, K contiguous): lane L (mm=L&15, hi=L>>4) holds
//     elems 0..7  = A[mm][k + hi*8 .. +7]      -> base + k
//     elems 8..15 = A[mm][k + 16 + hi*8 .. +7] -> base + k + 16
//  B^T row nn = output column, same K order as A:
//     elems 0..7  = Bt[nn][k + hi*8 .. +7]      -> base + k
//     elems 8..15 = Bt[nn][k + 16 + hi*8 .. +7] -> base + k + 16
template <int MT, int NT>
__device__ __forceinline__ void wave_mma_t(const bf16* __restrict__ A, long lda,
                                           const bf16* __restrict__ Bt, long ldb,
                                           long row0, long col0, int K, int lane,
                                           v8f (&cc)[MT][NT]) {
  const bf16* ap[MT];
  const bf16* bp[NT];
#pragma unroll
  for (int i = 0; i < MT; ++i)
    ap[i] = A + (row0 + i * 16 + (lane & 15)) * lda + ((lane >> 4) << 3);
#pragma unroll
  for (int j = 0; j < NT; ++j)
    bp[j] = Bt + (col0 + j * 16 + (lane & 15)) * ldb + ((lane >> 4) << 3);
  for (int k = 0; k < K; k += 32) {
    if (k + 32 < K) {
      __builtin_prefetch(ap[0] + k + 32, 0, 1);
      __builtin_prefetch(bp[0] + k + 32, 0, 1);
    }
    v16bf af[MT], bfrag[NT];
#pragma unroll
    for (int i = 0; i < MT; ++i) {
      FragU f;
      f.q[0] = *(const uint4*)(ap[i] + k);
      f.q[1] = *(const uint4*)(ap[i] + k + 16);
      af[i] = f.v;
    }
#pragma unroll
    for (int j = 0; j < NT; ++j) {
      FragU f;
      f.q[0] = *(const uint4*)(bp[j] + k);
      f.q[1] = *(const uint4*)(bp[j] + k + 16);
      bfrag[j] = f.v;
    }
#pragma unroll
    for (int i = 0; i < MT; ++i)
#pragma unroll
      for (int j = 0; j < NT; ++j)
        cc[i][j] = __builtin_amdgcn_wmma_f32_16x16x32_bf16(
            false, af[i], false, bfrag[j], (short)0, cc[i][j], false, false);
  }
}

template <int MT, int NT>
__device__ __forceinline__ void zero_acc_t(v8f (&cc)[MT][NT]) {
#pragma unroll
  for (int i = 0; i < MT; ++i)
#pragma unroll
    for (int j = 0; j < NT; ++j)
#pragma unroll
      for (int r = 0; r < 8; ++r) cc[i][j][r] = 0.0f;
}

// ---------------------------------------------------------------------------
// Weight GEMMs: 128 threads = 4 waves (2x2), wave tile 64x32 -> block 128x64
// ---------------------------------------------------------------------------

// fc1: C = relu(A @ Bt^T + bias), f32 out
__global__ __launch_bounds__(128) void gemm_bias_relu(
    const bf16* __restrict__ A, long lda, const bf16* __restrict__ Bt, long ldb,
    const float* __restrict__ bias, float* __restrict__ C, long ldc, int K) {
  int lane = threadIdx.x & 31, w = threadIdx.x >> 5;
  long row0 = (long)blockIdx.y * 128 + (w >> 1) * 64;
  long col0 = (long)blockIdx.x * 64 + (w & 1) * 32;
  v8f cc[4][2]; zero_acc_t(cc);
  wave_mma_t(A, lda, Bt, ldb, row0, col0, K, lane, cc);
  int nn = lane & 15, hi = lane >> 4;
#pragma unroll
  for (int ti = 0; ti < 4; ++ti)
#pragma unroll
    for (int tj = 0; tj < 2; ++tj)
#pragma unroll
      for (int r = 0; r < 8; ++r) {
        long row = row0 + ti * 16 + hi * 8 + r;
        long col = col0 + tj * 16 + nn;
        C[row * ldc + col] = fmaxf(cc[ti][tj][r] + bias[col], 0.0f);
      }
}

// qkv: epilogue splits into q (bf16, * d^-0.5), k (bf16), v (f32), head-major
__global__ __launch_bounds__(128) void gemm_qkv_split(
    const bf16* __restrict__ A, const bf16* __restrict__ Bt,
    const float* __restrict__ bias, bf16* __restrict__ qb,
    bf16* __restrict__ kb, float* __restrict__ vb) {
  int lane = threadIdx.x & 31, w = threadIdx.x >> 5;
  long row0 = (long)blockIdx.y * 128 + (w >> 1) * 64;
  long col0 = (long)blockIdx.x * 64 + (w & 1) * 32;
  v8f cc[4][2]; zero_acc_t(cc);
  wave_mma_t(A, 512, Bt, 512, row0, col0, 512, lane, cc);
  int nn = lane & 15, hi = lane >> 4;
#pragma unroll
  for (int ti = 0; ti < 4; ++ti)
#pragma unroll
    for (int tj = 0; tj < 2; ++tj)
#pragma unroll
      for (int r = 0; r < 8; ++r) {
        long row = row0 + ti * 16 + hi * 8 + r;         // 0..32767 = b*16384+i
        int col = (int)(col0 + tj * 16 + nn);           // 0..1535
        float val = cc[ti][tj][r] + bias[col];
        int b = (int)(row >> 14), i = (int)(row & 16383);
        int which = col >> 9, head = (col >> 6) & 7, dd = col & 63;
        long o = (((long)b * 8 + head) * 16384 + i) * 64 + dd;
        if (which == 0)      qb[o] = (bf16)(val * 0.125f);   // d^-0.5, d=64
        else if (which == 1) kb[o] = (bf16)val;
        else                 vb[o] = val;
      }
}

// out projection: res[b, i-254, col] += C + bias  (only tokens i>=254 survive)
__global__ __launch_bounds__(128) void gemm_out_res(
    const bf16* __restrict__ A, const bf16* __restrict__ Bt,
    const float* __restrict__ bias, float* __restrict__ res) {
  int lane = threadIdx.x & 31, w = threadIdx.x >> 5;
  long row0 = (long)blockIdx.y * 128 + (w >> 1) * 64;
  long col0 = (long)blockIdx.x * 64 + (w & 1) * 32;
  v8f cc[4][2]; zero_acc_t(cc);
  wave_mma_t(A, 512, Bt, 512, row0, col0, 512, lane, cc);
  int nn = lane & 15, hi = lane >> 4;
#pragma unroll
  for (int ti = 0; ti < 4; ++ti)
#pragma unroll
    for (int tj = 0; tj < 2; ++tj)
#pragma unroll
      for (int r = 0; r < 8; ++r) {
        long row = row0 + ti * 16 + hi * 8 + r;
        long col = col0 + tj * 16 + nn;
        int b = (int)(row >> 14), i = (int)(row & 16383);
        if (i >= 254) {
          long o = ((long)b * 16130 + (i - 254)) * 512 + col;
          res[o] += cc[ti][tj][r] + bias[col];
        }
      }
}

// ---------------------------------------------------------------------------
// Batched attention GEMMs: 128 threads = 4 waves (2x2), wave tile 32x32
// ---------------------------------------------------------------------------

__global__ __launch_bounds__(128) void gemm_bat_f32(
    const bf16* __restrict__ A, long lda, long sA,
    const bf16* __restrict__ Bt, long ldb, long sB,
    float* __restrict__ C, long ldc, long sC, int K) {
  int lane = threadIdx.x & 31, w = threadIdx.x >> 5;
  long row0 = (long)blockIdx.y * 64 + (w >> 1) * 32;
  long col0 = (long)blockIdx.x * 64 + (w & 1) * 32;
  const bf16* Az = A + (long)blockIdx.z * sA;
  const bf16* Bz = Bt + (long)blockIdx.z * sB;
  float* Cz = C + (long)blockIdx.z * sC;
  v8f cc[2][2]; zero_acc_t(cc);
  wave_mma_t(Az, lda, Bz, ldb, row0, col0, K, lane, cc);
  int nn = lane & 15, hi = lane >> 4;
#pragma unroll
  for (int ti = 0; ti < 2; ++ti)
#pragma unroll
    for (int tj = 0; tj < 2; ++tj)
#pragma unroll
      for (int r = 0; r < 8; ++r) {
        long row = row0 + ti * 16 + hi * 8 + r;
        long col = col0 + tj * 16 + nn;
        Cz[row * ldc + col] = cc[ti][tj][r];
      }
}

// ---------------------------------------------------------------------------
// Elementwise / reduction kernels (fp32 VALU, bandwidth-trivial)
// ---------------------------------------------------------------------------

__global__ void cvt_bf16(const float* __restrict__ in, bf16* __restrict__ out, long n) {
  long i = (long)blockIdx.x * 256 + threadIdx.x;
  if (i < n) out[i] = (bf16)in[i];
}

// build token buffer h[B,16130,512]: token0=cls, 1..16000 = fc1 rows, dup tail
__global__ void assemble_h(const float* __restrict__ c0, const float* __restrict__ cls,
                           float* __restrict__ h) {
  long idx = (long)blockIdx.x * 256 + threadIdx.x;
  if (idx >= 2L * 16130 * 512) return;
  int c = (int)(idx & 511);
  long rt = idx >> 9;
  int b = (int)(rt / 16130), t = (int)(rt % 16130);
  float val;
  if (t == 0) val = cls[c];
  else {
    int src = (t <= 16000) ? (t - 1) : (t - 1 - 16000);
    val = c0[((long)b * 16000 + src) * 512 + c];
  }
  h[idx] = val;
}

// layernorm token -> bf16 row of xn[B,16384,512]; first 254 rows zero-pad
__global__ void ln_pad_bf16(const float* __restrict__ res, const float* __restrict__ w,
                            const float* __restrict__ b, bf16* __restrict__ out) {
  int row = blockIdx.x;                 // b*16384 + i
  int bi = row >> 14, i = row & 16383;
  int tid = threadIdx.x;
  bf16* o = out + (long)row * 512;
  if (i < 254) { o[tid] = (bf16)0.0f; o[tid + 256] = (bf16)0.0f; return; }
  const float* x = res + ((long)bi * 16130 + (i - 254)) * 512;
  __shared__ float sred[256];
  float x0 = x[tid], x1 = x[tid + 256];
  sred[tid] = x0 + x1; __syncthreads();
  for (int s = 128; s > 0; s >>= 1) { if (tid < s) sred[tid] += sred[tid + s]; __syncthreads(); }
  float mu = sred[0] * (1.0f / 512.0f); __syncthreads();
  float d0 = x0 - mu, d1 = x1 - mu;
  sred[tid] = d0 * d0 + d1 * d1; __syncthreads();
  for (int s = 128; s > 0; s >>= 1) { if (tid < s) sred[tid] += sred[tid + s]; __syncthreads(); }
  float rstd = rsqrtf(sred[0] * (1.0f / 512.0f) + 1e-5f);
  o[tid]       = (bf16)(d0 * rstd * w[tid] + b[tid]);
  o[tid + 256] = (bf16)(d1 * rstd * w[tid + 256] + b[tid + 256]);
}

// landmark means: out[z,j,dd] = mean over 64 rows of in[z, j*64+t, dd]
__global__ void landmark_mean(const bf16* __restrict__ in, bf16* __restrict__ out) {
  int idx = blockIdx.x * 256 + threadIdx.x;  // 16*256*64
  if (idx >= 16 * 256 * 64) return;
  int dd = idx & 63, j = (idx >> 6) & 255, z = idx >> 14;
  const bf16* p = in + (((long)z * 16384) + (long)j * 64) * 64 + dd;
  float s = 0.0f;
  for (int t = 0; t < 64; ++t) s += (float)p[(long)t * 64];
  out[idx] = (bf16)(s * (1.0f / 64.0f));
}

__global__ void v_to_vT(const float* __restrict__ v, bf16* __restrict__ vT) {
  long idx = (long)blockIdx.x * 256 + threadIdx.x;   // 16*16384*64
  if (idx >= 16L * 16384 * 64) return;
  int dd = (int)(idx & 63);
  long zi = idx >> 6;
  int i = (int)(zi & 16383), z = (int)(zi >> 14);
  vT[((long)z * 64 + dd) * 16384 + i] = (bf16)v[idx];
}

// row softmax in place (f32), optional bf16 mirror
__global__ void softmax_rows(float* __restrict__ X, bf16* __restrict__ Y, int W) {
  long row = blockIdx.x;
  float* x = X + row * (long)W;
  __shared__ float sred[256];
  int tid = threadIdx.x;
  float m = -1e30f;
  for (int j = tid; j < W; j += 256) m = fmaxf(m, x[j]);
  sred[tid] = m; __syncthreads();
  for (int s = 128; s > 0; s >>= 1) { if (tid < s) sred[tid] = fmaxf(sred[tid], sred[tid + s]); __syncthreads(); }
  m = sred[0]; __syncthreads();
  float sum = 0.0f;
  for (int j = tid; j < W; j += 256) { float e = __expf(x[j] - m); x[j] = e; sum += e; }
  sred[tid] = sum; __syncthreads();
  for (int s = 128; s > 0; s >>= 1) { if (tid < s) sred[tid] += sred[tid + s]; __syncthreads(); }
  float inv = 1.0f / sred[0];
  for (int j = tid; j < W; j += 256) {
    float p = x[j] * inv;
    x[j] = p;
    if (Y) Y[row * (long)W + j] = (bf16)p;
  }
}

// pinv: per-z max row/col abs sums
__global__ void pinv_norms(const float* __restrict__ a, float* __restrict__ norms) {
  int z = blockIdx.x, tid = threadIdx.x;
  const float* A = a + (long)z * 65536;
  __shared__ float srow[256], scol[256];
  float rs = 0.0f, cs = 0.0f;
  for (int j = 0; j < 256; ++j) { rs += fabsf(A[tid * 256 + j]); cs += fabsf(A[j * 256 + tid]); }
  srow[tid] = rs; scol[tid] = cs; __syncthreads();
  for (int s = 128; s > 0; s >>= 1) {
    if (tid < s) { srow[tid] = fmaxf(srow[tid], srow[tid + s]); scol[tid] = fmaxf(scol[tid], scol[tid + s]); }
    __syncthreads();
  }
  if (tid == 0) { norms[z * 2] = srow[0]; norms[z * 2 + 1] = scol[0]; }
}

// z0 = a^T / (global max rowsum * global max colsum)
__global__ void pinv_transpose_scale(const float* __restrict__ a, const float* __restrict__ norms,
                                     float* __restrict__ z0) {
  int z = blockIdx.x, tid = threadIdx.x;
  float mr = 0.0f, mc = 0.0f;
  for (int t = 0; t < 16; ++t) { mr = fmaxf(mr, norms[t * 2]); mc = fmaxf(mc, norms[t * 2 + 1]); }
  float inv = 1.0f / (mr * mc);
  const float* A = a + (long)z * 65536;
  float* Z = z0 + (long)z * 65536;
  for (int j = 0; j < 256; ++j) Z[tid * 256 + j] = A[j * 256 + tid] * inv;
}

// batched 256x256x256 fp32 matmul: C = alpha * A@B
__global__ void mm256(const float* __restrict__ A, const float* __restrict__ B,
                      float* __restrict__ C, float alpha) {
  long idx = (long)blockIdx.x * 256 + threadIdx.x;   // 16*65536
  if (idx >= 16L * 65536) return;
  int j = (int)(idx & 255), i = (int)((idx >> 8) & 255), z = (int)(idx >> 16);
  const float* Ar = A + (long)z * 65536 + i * 256;
  const float* Bc = B + (long)z * 65536 + j;
  float s = 0.0f;
  for (int k = 0; k < 256; ++k) s += Ar[k] * Bc[k * 256];
  C[idx] = alpha * s;
}

__global__ void aI_minus(const float* __restrict__ A, float* __restrict__ C, float alpha) {
  long idx = (long)blockIdx.x * 256 + threadIdx.x;
  if (idx >= 16L * 65536) return;
  int j = (int)(idx & 255), i = (int)((idx >> 8) & 255);
  C[idx] = ((i == j) ? alpha : 0.0f) - A[idx];
}

// pk = pinv_z @ kv ; store transposed bf16 pkT[z, dd, m]
__global__ void pk_to_pkT(const float* __restrict__ Zm, const float* __restrict__ kv,
                          bf16* __restrict__ pkT) {
  int idx = blockIdx.x * 256 + threadIdx.x;   // 16*256*64
  if (idx >= 16 * 16384) return;
  int dd = idx & 63, i = (idx >> 6) & 255, z = idx >> 14;
  const float* Zr = Zm + (long)z * 65536 + i * 256;
  const float* Kc = kv + (long)z * 16384 + dd;
  float s = 0.0f;
  for (int k = 0; k < 256; ++k) s += Zr[k] * Kc[k * 64];
  pkT[((long)z * 64 + dd) * 256 + i] = (bf16)s;
}

// out_heads += depthwise conv over sequence (33 taps, pad 16)
__global__ void dwconv_add(const float* __restrict__ v, const float* __restrict__ rw,
                           float* __restrict__ outh) {
  long idx = (long)blockIdx.x * 256 + threadIdx.x;   // 16*16384*64
  if (idx >= 16L * 16384 * 64) return;
  long zi = idx >> 6;
  int i = (int)(zi & 16383), z = (int)(zi >> 14);
  const float* w = rw + (z & 7) * 33;
  float s = 0.0f;
  for (int t = 0; t < 33; ++t) {
    int ii = i + t - 16;
    if (ii >= 0 && ii < 16384) s += v[idx + (long)(t - 16) * 64] * w[t];
  }
  outh[idx] += s;
}

__global__ void merge_heads(const float* __restrict__ outh, bf16* __restrict__ A2) {
  long idx = (long)blockIdx.x * 256 + threadIdx.x;   // 2*16384*512
  if (idx >= 2L * 16384 * 512) return;
  int c = (int)(idx & 511);
  long bi = idx >> 9;
  int i = (int)(bi & 16383), b = (int)(bi >> 14);
  int head = c >> 6, dd = c & 63;
  A2[idx] = (bf16)outh[(((long)(b * 8 + head)) * 16384 + i) * 64 + dd];
}

// PPEG: h2 = identity + dwconv7 + b7 + dwconv5 + b5 + dwconv3 + b3 (127x127)
__global__ void ppeg_kernel(const float* __restrict__ h,
                            const float* __restrict__ w7, const float* __restrict__ b7,
                            const float* __restrict__ w5, const float* __restrict__ b5,
                            const float* __restrict__ w3, const float* __restrict__ b3,
                            float* __restrict__ h2) {
  long idx = (long)blockIdx.x * 256 + threadIdx.x;   // 2*16130*512
  if (idx >= 2L * 16130 * 512) return;
  int c = (int)(idx & 511);
  long bt = idx >> 9;
  int t = (int)(bt % 16130), b = (int)(bt / 16130);
  if (t == 0) { h2[idx] = h[idx]; return; }
  int pos = t - 1, y = pos / 127, x = pos % 127;
  const float* base = h + ((long)b * 16130 + 1) * 512 + c;
  float acc = base[(long)pos * 512] + b7[c] + b5[c] + b3[c];
  for (int ky = 0; ky < 7; ++ky) {
    int yy = y + ky - 3; if (yy < 0 || yy >= 127) continue;
    for (int kx = 0; kx < 7; ++kx) {
      int xx = x + kx - 3; if (xx < 0 || xx >= 127) continue;
      acc += base[(long)(yy * 127 + xx) * 512] * w7[c * 49 + ky * 7 + kx];
    }
  }
  for (int ky = 0; ky < 5; ++ky) {
    int yy = y + ky - 2; if (yy < 0 || yy >= 127) continue;
    for (int kx = 0; kx < 5; ++kx) {
      int xx = x + kx - 2; if (xx < 0 || xx >= 127) continue;
      acc += base[(long)(yy * 127 + xx) * 512] * w5[c * 25 + ky * 5 + kx];
    }
  }
  for (int ky = 0; ky < 3; ++ky) {
    int yy = y + ky - 1; if (yy < 0 || yy >= 127) continue;
    for (int kx = 0; kx < 3; ++kx) {
      int xx = x + kx - 1; if (xx < 0 || xx >= 127) continue;
      acc += base[(long)(yy * 127 + xx) * 512] * w3[c * 9 + ky * 3 + kx];
    }
  }
  h2[idx] = acc;
}

// final layernorm on cls token + fc2 (2 outputs per batch)
__global__ void final_head(const float* __restrict__ res, const float* __restrict__ nw,
                           const float* __restrict__ nb, const float* __restrict__ fw,
                           const float* __restrict__ fb, float* __restrict__ out) {
  int b = blockIdx.x, tid = threadIdx.x;
  const float* x = res + (long)b * 16130 * 512;
  __shared__ float sred[256];
  float x0 = x[tid], x1 = x[tid + 256];
  sred[tid] = x0 + x1; __syncthreads();
  for (int s = 128; s > 0; s >>= 1) { if (tid < s) sred[tid] += sred[tid + s]; __syncthreads(); }
  float mu = sred[0] * (1.0f / 512.0f); __syncthreads();
  float d0 = x0 - mu, d1 = x1 - mu;
  sred[tid] = d0 * d0 + d1 * d1; __syncthreads();
  for (int s = 128; s > 0; s >>= 1) { if (tid < s) sred[tid] += sred[tid + s]; __syncthreads(); }
  float rstd = rsqrtf(sred[0] * (1.0f / 512.0f) + 1e-5f);
  float y0 = d0 * rstd * nw[tid] + nb[tid];
  float y1 = d1 * rstd * nw[tid + 256] + nb[tid + 256];
  __syncthreads();
  for (int cc = 0; cc < 2; ++cc) {
    sred[tid] = y0 * fw[cc * 512 + tid] + y1 * fw[cc * 512 + tid + 256];
    __syncthreads();
    for (int s = 128; s > 0; s >>= 1) { if (tid < s) sred[tid] += sred[tid + s]; __syncthreads(); }
    if (tid == 0) out[b * 2 + cc] = sred[0] + fb[cc];
    __syncthreads();
  }
}

// ---------------------------------------------------------------------------
// Host orchestration
// ---------------------------------------------------------------------------

static inline unsigned gdiv(long n, int blk) { return (unsigned)((n + blk - 1) / blk); }

extern "C" void kernel_launch(void* const* d_in, const int* in_sizes, int n_in,
                              void* d_out, int out_size, void* d_ws, size_t ws_size,
                              hipStream_t stream) {
  (void)in_sizes; (void)n_in; (void)out_size; (void)ws_size;
  const float* x          = (const float*)d_in[0];
  const float* fc1_w      = (const float*)d_in[1];
  const float* fc1_b      = (const float*)d_in[2];
  const float* cls_token  = (const float*)d_in[3];
  const float* ln_w       = (const float*)d_in[4];
  const float* ln_b       = (const float*)d_in[5];
  const float* qkv_w      = (const float*)d_in[6];
  const float* qkv_b      = (const float*)d_in[7];
  const float* attn_out_w = (const float*)d_in[8];
  const float* attn_out_b = (const float*)d_in[9];
  const float* res_conv_w = (const float*)d_in[10];
  const float* w7 = (const float*)d_in[11]; const float* b7 = (const float*)d_in[12];
  const float* w5 = (const float*)d_in[13]; const float* b5 = (const float*)d_in[14];
  const float* w3 = (const float*)d_in[15]; const float* b3 = (const float*)d_in[16];
  const float* norm_w = (const float*)d_in[17]; const float* norm_b = (const float*)d_in[18];
  const float* fc2_w = (const float*)d_in[19]; const float* fc2_b = (const float*)d_in[20];
  float* out = (float*)d_out;

  // bump allocator on d_ws (total ~1.0 GB)
  char* ws = (char*)d_ws;
  size_t off = 0;
  auto alloc = [&](size_t bytes) -> void* {
    off = (off + 255) & ~(size_t)255;
    void* p = ws + off;
    off += bytes;
    return p;
  };
  bf16* xbf    = (bf16*)alloc(32768000L * 2);
  bf16* fc1wbf = (bf16*)alloc(524288L * 2);
  bf16* qkvwbf = (bf16*)alloc(1572864L * 2);
  bf16* outwbf = (bf16*)alloc(524288L * 2);
  float* c0    = (float*)alloc(32000L * 512 * 4);
  float* h     = (float*)alloc(2L * 16130 * 512 * 4);
  float* h2    = (float*)alloc(2L * 16130 * 512 * 4);
  bf16* xn     = (bf16*)alloc(2L * 16384 * 512 * 2);
  bf16* qb     = (bf16*)alloc(16777216L * 2);
  bf16* kb     = (bf16*)alloc(16777216L * 2);
  float* vb    = (float*)alloc(16777216L * 4);
  bf16* vT     = (bf16*)alloc(16777216L * 2);
  bf16* ql     = (bf16*)alloc(262144L * 2);
  bf16* kl     = (bf16*)alloc(262144L * 2);
  float* attn2 = (float*)alloc(16L * 65536 * 4);
  float* Za    = (float*)alloc(16L * 65536 * 4);
  float* Zb    = (float*)alloc(16L * 65536 * 4);
  float* T1    = (float*)alloc(16L * 65536 * 4);
  float* T2    = (float*)alloc(16L * 65536 * 4);
  float* T3    = (float*)alloc(16L * 65536 * 4);
  float* norms = (float*)alloc(32L * 4);
  float* kvb   = (float*)alloc(16L * 16384 * 4);
  bf16* pkT    = (bf16*)alloc(262144L * 2);
  float* P1    = (float*)alloc(16L * 256 * 16384 * 4);   // logits (f32)
  bf16* P2     = (bf16*)alloc(16L * 256 * 16384 * 2);    // probs (bf16)
  float* outh  = (float*)alloc(16777216L * 4);
  bf16* A2     = (bf16*)alloc(2L * 16384 * 512 * 2);

  // weight / input conversions to bf16
  cvt_bf16<<<gdiv(32768000, 256), 256, 0, stream>>>(x, xbf, 32768000L);
  cvt_bf16<<<gdiv(524288, 256), 256, 0, stream>>>(fc1_w, fc1wbf, 524288L);
  cvt_bf16<<<gdiv(1572864, 256), 256, 0, stream>>>(qkv_w, qkvwbf, 1572864L);
  cvt_bf16<<<gdiv(524288, 256), 256, 0, stream>>>(attn_out_w, outwbf, 524288L);

  // fc1 + relu, then assemble token buffer (cls + dup tail)
  gemm_bias_relu<<<dim3(512 / 64, 32000 / 128), 128, 0, stream>>>(
      xbf, 1024, fc1wbf, 1024, fc1_b, c0, 512, 1024);
  assemble_h<<<gdiv(2L * 16130 * 512, 256), 256, 0, stream>>>(c0, cls_token, h);

  for (int layer = 0; layer < 2; ++layer) {
    float* res = layer ? h2 : h;
    ln_pad_bf16<<<2 * 16384, 256, 0, stream>>>(res, ln_w + layer * 512, ln_b + layer * 512, xn);
    gemm_qkv_split<<<dim3(1536 / 64, 32768 / 128), 128, 0, stream>>>(
        xn, qkvwbf + (long)layer * 786432, qkv_b + layer * 1536, qb, kb, vb);
    landmark_mean<<<gdiv(262144, 256), 256, 0, stream>>>(qb, ql);
    landmark_mean<<<gdiv(262144, 256), 256, 0, stream>>>(kb, kl);
    v_to_vT<<<gdiv(16777216L, 256), 256, 0, stream>>>(vb, vT);

    // attn2 = softmax(q_l @ k_l^T) ; pinv via Newton-Schulz (fp32)
    gemm_bat_f32<<<dim3(4, 4, 16), 128, 0, stream>>>(ql, 64, 16384, kl, 64, 16384,
                                                     attn2, 256, 65536, 64);
    softmax_rows<<<4096, 256, 0, stream>>>(attn2, (bf16*)0, 256);
    pinv_norms<<<16, 256, 0, stream>>>(attn2, norms);
    pinv_transpose_scale<<<16, 256, 0, stream>>>(attn2, norms, Za);
    float* Zin = Za; float* Zout = Zb;
    for (int it = 0; it < 6; ++it) {
      unsigned g = gdiv(16L * 65536, 256);
      mm256<<<g, 256, 0, stream>>>(attn2, Zin, T1, 1.0f);        // az
      aI_minus<<<g, 256, 0, stream>>>(T1, T2, 7.0f);             // 7I - az
      mm256<<<g, 256, 0, stream>>>(T1, T2, T3, 1.0f);            // az @ (...)
      aI_minus<<<g, 256, 0, stream>>>(T3, T2, 15.0f);            // 15I - ...
      mm256<<<g, 256, 0, stream>>>(T1, T2, T3, 1.0f);            // az @ (...)
      aI_minus<<<g, 256, 0, stream>>>(T3, T2, 13.0f);            // 13I - ...
      mm256<<<g, 256, 0, stream>>>(Zin, T2, Zout, 0.25f);        // z = 0.25 z @ (...)
      float* t = Zin; Zin = Zout; Zout = t;
    }

    // attn3 = softmax(q_l @ k^T) ; kv = attn3 @ v ; pkT = (pinv @ kv)^T
    gemm_bat_f32<<<dim3(256, 4, 16), 128, 0, stream>>>(ql, 64, 16384, kb, 64, 16384L * 64,
                                                       P1, 16384, 256L * 16384, 64);
    softmax_rows<<<4096, 256, 0, stream>>>(P1, P2, 16384);
    gemm_bat_f32<<<dim3(1, 4, 16), 128, 0, stream>>>(P2, 16384, 256L * 16384, vT, 16384,
                                                     64L * 16384, kvb, 64, 16384, 16384);
    pk_to_pkT<<<gdiv(262144, 256), 256, 0, stream>>>(Zin, kvb, pkT);

    // attn1 = softmax(q @ k_l^T) ; out = attn1 @ pk (+ dwconv(v))
    gemm_bat_f32<<<dim3(4, 256, 16), 128, 0, stream>>>(qb, 64, 16384L * 64, kl, 64, 16384,
                                                       P1, 256, 16384L * 256, 64);
    softmax_rows<<<16 * 16384, 256, 0, stream>>>(P1, P2, 256);
    gemm_bat_f32<<<dim3(1, 256, 16), 128, 0, stream>>>(P2, 256, 16384L * 256, pkT, 256,
                                                       64L * 256, outh, 64, 16384L * 64, 256);
    dwconv_add<<<gdiv(16777216L, 256), 256, 0, stream>>>(vb, res_conv_w + layer * 264, outh);
    merge_heads<<<gdiv(2L * 16384 * 512, 256), 256, 0, stream>>>(outh, A2);
    gemm_out_res<<<dim3(512 / 64, 32768 / 128), 128, 0, stream>>>(
        A2, outwbf + (long)layer * 262144, attn_out_b + layer * 512, res);

    if (layer == 0)
      ppeg_kernel<<<gdiv(2L * 16130 * 512, 256), 256, 0, stream>>>(h, w7, b7, w5, b5, w3, b3, h2);
  }

  final_head<<<2, 256, 0, stream>>>(h2, norm_w, norm_b, fc2_w, fc2_b, out);
}
